// GGNN_85598698209315
// MI455X (gfx1250) — matrix operand order repaired and checked by reading, not compiled, not run on hardware
//
#include <hip/hip_runtime.h>
#include <hip/hip_bf16.h>

#define NNODES 50000
#define NEDGES 400000
#define NTYPES 4
#define DIN    64
#define DOUT   128
#define NGRAPH 64
#define NSTEPS 8

typedef __attribute__((ext_vector_type(16))) __bf16 v16bf;
typedef __attribute__((ext_vector_type(8)))  __bf16 v8bf;
typedef __attribute__((ext_vector_type(4)))  __bf16 v4bf;
typedef __attribute__((ext_vector_type(8)))  float  v8f;

// ---------------------------------------------------------------------------
// WMMA helper: D = A(16x32 bf16) * B(32x16 bf16) + C(f32)
// ---------------------------------------------------------------------------
__device__ __forceinline__ v8f bwmma(v16bf a, v16bf b, v8f c) {
    return __builtin_amdgcn_wmma_f32_16x16x32_bf16(
        /*neg_a=*/false, a, /*neg_b=*/false, b,
        /*c_mod=*/(short)0, c, /*reuse_a=*/false, /*reuse_b=*/false);
}

// A-operand loader, bf16 source, row-major [rows][128].
// CDNA5 16-bit A 16x32 layout: lane l -> M = l&15;
//   halves 0..7  = K = (l>>4)*8 + j        (VGPR 0..3)
//   halves 8..15 = K = 16 + (l>>4)*8 + j   (VGPR 4..7)
__device__ __forceinline__ v16bf load_a_bf16(const __bf16* base, int lane, int kt) {
    int r  = lane & 15;
    int ko = (lane >> 4) * 8;
    const __bf16* p = base + (size_t)r * DOUT + kt * 32 + ko;
    v8bf lo = *(const v8bf*)p;
    v8bf hi = *(const v8bf*)(p + 16);
    v16bf out;
#pragma unroll
    for (int i = 0; i < 8; ++i) { out[i] = lo[i]; out[i + 8] = hi[i]; }
    return out;
}

// Same, but from an fp32 source (aggregated messages), converting to bf16.
__device__ __forceinline__ v16bf load_a_f32(const float* base, int lane, int kt) {
    int r  = lane & 15;
    int ko = (lane >> 4) * 8;
    const float* p = base + (size_t)r * DOUT + kt * 32 + ko;
    v16bf out;
#pragma unroll
    for (int i = 0; i < 8; ++i) {
        out[i]     = (__bf16)p[i];
        out[i + 8] = (__bf16)p[16 + i];
    }
    return out;
}

// ---------------------------------------------------------------------------
// Pack an fp32 row-major [128 x ncols] weight into WMMA B-operand tiles.
// Tile (kt, nt), kt in 0..3 (K=128/32), nt in 0..NT-1 (N/16).
// B 32x16 bf16 layout: lane l -> N = l&15, halves j = K = (j>>3)*16 + (l>>4)*8 + (j&7).
// Packed element index: tile*512 + lane*16 + j, tile = kt*NT + nt.
// ---------------------------------------------------------------------------
__global__ void pack_b_kernel(const float* __restrict__ src, __bf16* __restrict__ dst,
                              int NT, int ncols) {
    int total = 4 * NT * 512;
    for (int i = blockIdx.x * blockDim.x + threadIdx.x; i < total;
         i += gridDim.x * blockDim.x) {
        int tile = i >> 9;
        int r    = i & 511;
        int lane = r >> 4, j = r & 15;
        int kt = tile / NT, nt = tile % NT;
        int k = kt * 32 + (j >> 3) * 16 + (lane >> 4) * 8 + (j & 7);
        int n = nt * 16 + (lane & 15);
        dst[i] = (__bf16)src[(size_t)k * ncols + n];
    }
}

// h = [features | 0], plus bf16 mirror
__global__ void init_h_kernel(const float* __restrict__ feat,
                              float* __restrict__ h, __bf16* __restrict__ hb) {
    int i = blockIdx.x * blockDim.x + threadIdx.x;
    if (i >= NNODES * DOUT) return;
    int n = i >> 7, c = i & 127;
    float v = (c < DIN) ? feat[(size_t)n * DIN + c] : 0.0f;
    h[i]  = v;
    hb[i] = (__bf16)v;
}

__global__ void zero_f32_kernel(float* __restrict__ p, int n) {
    for (int i = blockIdx.x * blockDim.x + threadIdx.x; i < n;
         i += gridDim.x * blockDim.x)
        p[i] = 0.0f;
}

// ---------------------------------------------------------------------------
// Message transform: tmsg[n, ty, :] = bf16( h[n,:] @ W_msg[ty] + b_msg[ty] )
// One wave per (nodeTile, etype): A operands (4 K-slices, 32 VGPRs) are loaded
// ONCE and reused across all 8 column tiles -> 32 WMMAs per wave, A-traffic /8.
// Waves total: (N/16) * T = 12500 -> 1563 blocks x 8 waves (guarded).
// ---------------------------------------------------------------------------
__global__ __launch_bounds__(256)
void msg_gemm_kernel(const __bf16* __restrict__ hb, const __bf16* __restrict__ wmsg_p,
                     const float* __restrict__ b_msg, __bf16* __restrict__ tmsg) {
    int wave = threadIdx.x >> 5, lane = threadIdx.x & 31;
    int tid  = blockIdx.x * 8 + wave;
    if (tid >= (NNODES / 16) * NTYPES) return;
    int nodeTile = tid >> 2;       // /NTYPES
    int ty       = tid & 3;

    const __bf16* arow = hb + (size_t)nodeTile * 16 * DOUT;
    v16bf a[4];
#pragma unroll
    for (int kt = 0; kt < 4; ++kt) a[kt] = load_a_bf16(arow, lane, kt);

    const __bf16* wbase = wmsg_p + (size_t)ty * 4 * 8 * 512;
    int colLane = lane & 15;
    int rbase   = nodeTile * 16 + (lane >> 4) * 8;

#pragma unroll
    for (int nt = 0; nt < 8; ++nt) {
        v8f acc = {};
#pragma unroll
        for (int kt = 0; kt < 4; ++kt) {
            v16bf b = *(const v16bf*)(wbase + (((size_t)kt * 8 + nt) * 512) + lane * 16);
            acc = bwmma(a[kt], b, acc);
        }
        int col    = nt * 16 + colLane;
        float bias = b_msg[ty * DOUT + col];
#pragma unroll
        for (int v = 0; v < 8; ++v) {
            float val = acc[v] + bias;
            tmsg[((size_t)(rbase + v) * NTYPES + ty) * DOUT + col] = (__bf16)val;
        }
    }
}

// ---------------------------------------------------------------------------
// Edge scatter: a[dst] += tmsg[src, etype].  One wave per edge, 4 cols/lane.
// tmsg is bf16 and L2-resident (51 MB << 192 MB L2); atomics stay in L2.
// ---------------------------------------------------------------------------
__global__ __launch_bounds__(256)
void scatter_kernel(const __bf16* __restrict__ tmsg, const int* __restrict__ src,
                    const int* __restrict__ dst, const int* __restrict__ ety,
                    float* __restrict__ a) {
    int wave = threadIdx.x >> 5, lane = threadIdx.x & 31;
    int e = blockIdx.x * 8 + wave;
    if (e >= NEDGES) return;
    int s = src[e], d = dst[e], t = ety[e];
    const __bf16* mp = tmsg + ((size_t)s * NTYPES + t) * DOUT + lane * 4;
    float* ap = a + (size_t)d * DOUT + lane * 4;
    v4bf m = *(const v4bf*)mp;
#pragma unroll
    for (int j = 0; j < 4; ++j) atomicAdd(ap + j, (float)m[j]);
}

// ---------------------------------------------------------------------------
// Fused GRU cell: one block (8 waves) per 16-node tile.
// All 8 A operands (4 K-slices x {agg, h}) are hoisted into registers and
// reused across the wave's 3 column tiles (24 WMMAs/wave, A-traffic /3).
// Gates computed from LDS: r,z,n -> h_new; writes h (f32) and hb (bf16).
// ---------------------------------------------------------------------------
__global__ __launch_bounds__(256)
void gru_kernel(const float* __restrict__ a, const __bf16* __restrict__ wih_p,
                const __bf16* __restrict__ whh_p, const float* __restrict__ b_ih,
                const float* __restrict__ b_hh, float* __restrict__ h,
                __bf16* __restrict__ hb) {
    __shared__ float gi[16 * 384];
    __shared__ float gh[16 * 384];
    int wave = threadIdx.x >> 5, lane = threadIdx.x & 31;
    int nodeTile = blockIdx.x;
    const float*  abase = a  + (size_t)nodeTile * 16 * DOUT;
    const __bf16* hbase = hb + (size_t)nodeTile * 16 * DOUT;

    v16bf av[4], hv[4];
#pragma unroll
    for (int kt = 0; kt < 4; ++kt) {
        av[kt] = load_a_f32(abase, lane, kt);
        hv[kt] = load_a_bf16(hbase, lane, kt);
    }

#pragma unroll
    for (int s = 0; s < 3; ++s) {
        int nt = wave + s * 8;              // col tile 0..23
        v8f ai = {}, ah = {};
#pragma unroll
        for (int kt = 0; kt < 4; ++kt) {
            v16bf bi = *(const v16bf*)(wih_p + (((size_t)kt * 24 + nt) * 512) + lane * 16);
            ai = bwmma(av[kt], bi, ai);
            v16bf bh = *(const v16bf*)(whh_p + (((size_t)kt * 24 + nt) * 512) + lane * 16);
            ah = bwmma(hv[kt], bh, ah);
        }
        int col   = nt * 16 + (lane & 15);
        int rbase = (lane >> 4) * 8;
#pragma unroll
        for (int v = 0; v < 8; ++v) {
            gi[(rbase + v) * 384 + col] = ai[v];
            gh[(rbase + v) * 384 + col] = ah[v];
        }
    }
    __syncthreads();

    int row = threadIdx.x >> 4;             // 0..15
    int c0  = (threadIdx.x & 15) * 8;       // 0..120
    size_t nidx = ((size_t)nodeTile * 16 + row) * DOUT;
#pragma unroll
    for (int j = 0; j < 8; ++j) {
        int c = c0 + j;
        float ir  = gi[row * 384 + c]       + b_ih[c];
        float iz  = gi[row * 384 + 128 + c] + b_ih[128 + c];
        float in_ = gi[row * 384 + 256 + c] + b_ih[256 + c];
        float hr  = gh[row * 384 + c]       + b_hh[c];
        float hz  = gh[row * 384 + 128 + c] + b_hh[128 + c];
        float hn  = gh[row * 384 + 256 + c] + b_hh[256 + c];
        float r   = 1.0f / (1.0f + __expf(-(ir + hr)));
        float z   = 1.0f / (1.0f + __expf(-(iz + hz)));
        float ng  = tanhf(in_ + r * hn);
        float hold = h[nidx + c];
        float hnew = (1.0f - z) * ng + z * hold;
        h[nidx + c]  = hnew;
        hb[nidx + c] = (__bf16)hnew;
    }
}

// ---------------------------------------------------------------------------
// Residual + per-graph segment sum. node2graph is sorted -> per-thread
// running sum, flush with atomicAdd only on graph-id change (~B flushes).
// Block covers 64 nodes; thread = (sub 0..1, col 0..127).
// ---------------------------------------------------------------------------
__global__ __launch_bounds__(256)
void segsum_kernel(const float* __restrict__ h, const float* __restrict__ feat,
                   const int* __restrict__ n2g, float* __restrict__ feats_g) {
    int c   = threadIdx.x & 127;
    int sub = threadIdx.x >> 7;
    int base = blockIdx.x * 64 + sub * 32;
    float sum = 0.0f;
    int gcur = -1;
    for (int i = 0; i < 32; ++i) {
        int n = base + i;
        if (n >= NNODES) break;
        int g = n2g[n];
        float v = h[(size_t)n * DOUT + c] +
                  ((c < DIN) ? feat[(size_t)n * DIN + c] : 0.0f);
        if (g != gcur) {
            if (gcur >= 0) atomicAdd(&feats_g[gcur * DOUT + c], sum);
            gcur = g; sum = v;
        } else {
            sum += v;
        }
    }
    if (gcur >= 0) atomicAdd(&feats_g[gcur * DOUT + c], sum);
}

// out[b, j] = feats_g[b,:] @ W_cls[:, j] + b_cls[j]   (64x2 outputs)
__global__ void cls_kernel(const float* __restrict__ feats_g,
                           const float* __restrict__ Wc,
                           const float* __restrict__ bc, float* __restrict__ out) {
    int t = threadIdx.x;
    if (t >= NGRAPH * 2) return;
    int b = t >> 1, j = t & 1;
    float s = bc[j];
#pragma unroll 8
    for (int d = 0; d < DOUT; ++d) s += feats_g[b * DOUT + d] * Wc[d * 2 + j];
    out[b * 2 + j] = s;
}

// ---------------------------------------------------------------------------
extern "C" void kernel_launch(void* const* d_in, const int* in_sizes, int n_in,
                              void* d_out, int out_size, void* d_ws, size_t ws_size,
                              hipStream_t stream) {
    (void)in_sizes; (void)n_in; (void)out_size; (void)ws_size;
    const float* features = (const float*)d_in[0];
    const int*   src      = (const int*)  d_in[1];
    const int*   dst      = (const int*)  d_in[2];
    const int*   etype    = (const int*)  d_in[3];
    const int*   n2g      = (const int*)  d_in[4];
    const float* W_msg    = (const float*)d_in[5];
    const float* b_msg    = (const float*)d_in[6];
    const float* w_ih     = (const float*)d_in[7];
    const float* b_ih     = (const float*)d_in[8];
    const float* w_hh     = (const float*)d_in[9];
    const float* b_hh     = (const float*)d_in[10];
    const float* W_cls    = (const float*)d_in[11];
    const float* b_cls    = (const float*)d_in[12];
    float* out = (float*)d_out;

    char* ws = (char*)d_ws;
    size_t off = 0;
    __bf16* hb     = (__bf16*)(ws + off); off += (size_t)NNODES * DOUT * 2;          // 12.8 MB
    float*  h      = (float*) (ws + off); off += (size_t)NNODES * DOUT * 4;          // 25.6 MB
    float*  agg    = (float*) (ws + off); off += (size_t)NNODES * DOUT * 4;          // 25.6 MB
    __bf16* tmsg   = (__bf16*)(ws + off); off += (size_t)NNODES * NTYPES * DOUT * 2; // 51.2 MB
    __bf16* wmsg_p = (__bf16*)(ws + off); off += (size_t)NTYPES * DOUT * DOUT * 2;   // 128 KB
    __bf16* wih_p  = (__bf16*)(ws + off); off += (size_t)DOUT * 384 * 2;             // 96 KB
    __bf16* whh_p  = (__bf16*)(ws + off); off += (size_t)DOUT * 384 * 2;             // 96 KB
    float*  fg     = (float*) (ws + off); off += (size_t)NGRAPH * DOUT * 4;          // 32 KB

    // 1) pack weights into WMMA B-operand layout (bf16)
    for (int t = 0; t < NTYPES; ++t)
        pack_b_kernel<<<64, 256, 0, stream>>>(W_msg + (size_t)t * DOUT * DOUT,
                                              wmsg_p + (size_t)t * DOUT * DOUT,
                                              /*NT=*/8, /*ncols=*/DOUT);
    pack_b_kernel<<<96, 256, 0, stream>>>(w_ih, wih_p, /*NT=*/24, /*ncols=*/384);
    pack_b_kernel<<<96, 256, 0, stream>>>(w_hh, whh_p, /*NT=*/24, /*ncols=*/384);

    // 2) h = [features | 0]
    init_h_kernel<<<(NNODES * DOUT + 255) / 256, 256, 0, stream>>>(features, h, hb);

    // 3) propagation steps
    const int msgWaves  = (NNODES / 16) * NTYPES;          // 12500
    const int msgBlocks = (msgWaves + 7) / 8;              // 1563
    for (int step = 0; step < NSTEPS; ++step) {
        zero_f32_kernel<<<4096, 256, 0, stream>>>(agg, NNODES * DOUT);
        msg_gemm_kernel<<<msgBlocks, 256, 0, stream>>>(hb, wmsg_p, b_msg, tmsg);
        scatter_kernel<<<(NEDGES + 7) / 8, 256, 0, stream>>>(tmsg, src, dst, etype, agg);
        gru_kernel<<<NNODES / 16, 256, 0, stream>>>(agg, wih_p, whh_p, b_ih, b_hh, h, hb);
    }

    // 4) readout
    zero_f32_kernel<<<32, 256, 0, stream>>>(fg, NGRAPH * DOUT);
    segsum_kernel<<<(NNODES + 63) / 64, 256, 0, stream>>>(h, features, n2g, fg);
    cls_kernel<<<1, 128, 0, stream>>>(fg, W_cls, b_cls, out);
}
